// MARLCommunicationLayer_25013889532569
// MI455X (gfx1250) — matrix operand order repaired and checked by reading, not compiled, not run on hardware
//
#include <hip/hip_runtime.h>
#include <hip/hip_bf16.h>
#include <stdint.h>

// ---------------------------------------------------------------------------
// MARL communication layer for MI455X (gfx1250, wave32, WMMA).
// Pipeline:
//  prep -> topk(sim, WMMA-fused) -> adjacency bitset -> 2x GAT (WMMA GEMM +
//  sparse masked-softmax aggregate + LN) -> enc/dec MLP GEMMs -> qkv GEMM ->
//  flash attention (WMMA QK^T + WMMA PV, online softmax) -> out-proj GEMM ->
//  gate -> final GEMM with residual.
// All matmul work uses v_wmma_f32_16x16x32_bf16 (f32 accumulate).
// GEMMs: 16x64 tile per wave (4 independent accumulators to fill the matrix
// pipe) + register double-buffering of fragments to hide load latency.
// ---------------------------------------------------------------------------

#define NAGENT 4096
#define DIM    256
#define NHEAD  4
#define HDIM   64
#define KTOP   5
#define NBR_CAP 1024

typedef __attribute__((ext_vector_type(16))) __bf16 v16bf;
typedef __attribute__((ext_vector_type(8)))  __bf16 v8bf;
typedef __attribute__((ext_vector_type(8)))  float  v8f;

static __device__ __forceinline__ __bf16 f2bf(float f) {
  unsigned u = __builtin_bit_cast(unsigned, f);
  u += 0x7fffu + ((u >> 16) & 1u);               // round-to-nearest-even
  unsigned short h = (unsigned short)(u >> 16);
  return __builtin_bit_cast(__bf16, h);
}

// A fragment: 16(M) x 32(K) bf16, row-major source with leading dim ld.
// Lane l holds row (l&15); lanes 0-15 K=0..15, lanes 16-31 K=16..31.
static __device__ __forceinline__ v16bf load_fragA(const __bf16* __restrict__ base,
                                                   int ld, int lane) {
  const int r = lane & 15, hi = (lane >> 4) & 1;
  const __bf16* p = base + (size_t)r * ld + hi * 16;
  v8bf lo = *(const v8bf*)p;
  v8bf hp = *(const v8bf*)(p + 8);
  return __builtin_shufflevector(lo, hp, 0,1,2,3,4,5,6,7,8,9,10,11,12,13,14,15);
}

// B fragment: 32(K) x 16(N) bf16 from B^T stored [N][K] row-major (ld = K-stride).
// Lane l holds column (l&15); lanes 0-15 K=0..15, lanes 16-31 K=16..31.
static __device__ __forceinline__ v16bf load_fragB(const __bf16* __restrict__ base,
                                                   int ld, int lane) {
  const int n = lane & 15, hi = (lane >> 4) & 1;
  const __bf16* p = base + (size_t)n * ld + hi * 16;
  v8bf lo = *(const v8bf*)p;
  v8bf hp = *(const v8bf*)(p + 8);
  return __builtin_shufflevector(lo, hp, 0,1,2,3,4,5,6,7,8,9,10,11,12,13,14,15);
}

static __device__ __forceinline__ v8f wmma_bf16(v16bf a, v16bf b, v8f c) {
  return __builtin_amdgcn_wmma_f32_16x16x32_bf16(false, a, false, b, (short)0, c,
                                                 false, false);
}

// ---------------------------------------------------------------------------
// prep: enhanced = states + agent_emb + tile(role,4); row-normalize for sim.
// ---------------------------------------------------------------------------
__global__ __launch_bounds__(256)
void prep_kernel(const float* __restrict__ states, const float* __restrict__ emb,
                 const float* __restrict__ role, float* __restrict__ xA,
                 __bf16* __restrict__ xA_bf, __bf16* __restrict__ ns_bf,
                 __bf16* __restrict__ states_bf) {
  const int i = blockIdx.x, c = threadIdx.x;
  const float s = states[(size_t)i * DIM + c];
  const float e = s + emb[(size_t)i * DIM + c] + role[(size_t)i * 64 + (c & 63)];
  __shared__ float red[256];
  red[c] = e * e;
  __syncthreads();
  for (int st = 128; st > 0; st >>= 1) {
    if (c < st) red[c] += red[c + st];
    __syncthreads();
  }
  const float inv = 1.f / fmaxf(sqrtf(red[0]), 1e-12f);
  xA[(size_t)i * DIM + c]        = e;
  xA_bf[(size_t)i * DIM + c]     = f2bf(e);
  ns_bf[(size_t)i * DIM + c]     = f2bf(e * inv);
  states_bf[(size_t)i * DIM + c] = f2bf(s);
}

// ---------------------------------------------------------------------------
// transpose+cast a f32 [R][C] weight into bf16 [C][R] (TN GEMM B operand).
// ---------------------------------------------------------------------------
__global__ void transpose_cast_kernel(const float* __restrict__ src,
                                      __bf16* __restrict__ dst, int R, int C) {
  const int idx = blockIdx.x * blockDim.x + threadIdx.x;
  if (idx >= R * C) return;
  const int r = idx / C, c = idx % C;
  dst[(size_t)c * R + r] = f2bf(src[idx]);
}

// ---------------------------------------------------------------------------
// Generic TN GEMM: C[MxN] = A(bf16,[M][K]) * B^T(bf16,[N][K]) with epilogue.
// One wave per 16 x (16*NT) tile: NT independent accumulators keep the matrix
// pipe busy; next-K fragments are preloaded into registers (double buffering)
// so s_wait_loadcnt covers loads issued a full K-step earlier.
// flags bit0 = ReLU. residual/Cbf/bias optional.
// ---------------------------------------------------------------------------
template <int NT>
__global__ __launch_bounds__(32)
void gemm_bf16_wmma_kernel(const __bf16* __restrict__ A, int lda,
                           const __bf16* __restrict__ Bt, int ldb,
                           float* __restrict__ C, __bf16* __restrict__ Cbf, int ldc,
                           int K, const float* __restrict__ bias,
                           const float* __restrict__ residual, int flags) {
  const int lane = threadIdx.x;
  const int m0 = blockIdx.y * 16, n0 = blockIdx.x * (16 * NT);
  const v8f z = {0.f, 0.f, 0.f, 0.f, 0.f, 0.f, 0.f, 0.f};
  v8f acc[NT];
#pragma unroll
  for (int t = 0; t < NT; ++t) acc[t] = z;
  const __bf16* ap = A + (size_t)m0 * lda;
  const __bf16* bp = Bt + (size_t)n0 * ldb;

  v16bf a = load_fragA(ap, lda, lane);
  v16bf b[NT];
#pragma unroll
  for (int t = 0; t < NT; ++t)
    b[t] = load_fragB(bp + (size_t)t * 16 * ldb, ldb, lane);

  for (int k = 0; k < K; k += 32) {
    v16bf an = a;
    v16bf bn[NT];
#pragma unroll
    for (int t = 0; t < NT; ++t) bn[t] = b[t];
    const int kn = k + 32;
    if (kn < K) {  // prefetch next-K fragments into registers
      an = load_fragA(ap + kn, lda, lane);
#pragma unroll
      for (int t = 0; t < NT; ++t)
        bn[t] = load_fragB(bp + (size_t)t * 16 * ldb + kn, ldb, lane);
    }
#pragma unroll
    for (int t = 0; t < NT; ++t) acc[t] = wmma_bf16(a, b[t], acc[t]);
    a = an;
#pragma unroll
    for (int t = 0; t < NT; ++t) b[t] = bn[t];
  }

  const int hi = lane >> 4, nn = lane & 15;
#pragma unroll
  for (int t = 0; t < NT; ++t) {
    const int col = n0 + t * 16 + nn;
    const float bv = bias ? bias[col] : 0.f;
#pragma unroll
    for (int r = 0; r < 8; ++r) {
      const int row = m0 + r + hi * 8;
      float v = acc[t][r] + bv;
      if (flags & 1) v = fmaxf(v, 0.f);
      if (residual) v += residual[(size_t)row * ldc + col];
      C[(size_t)row * ldc + col] = v;
      if (Cbf) Cbf[(size_t)row * ldc + col] = f2bf(v);
    }
  }
}

// ---------------------------------------------------------------------------
// Fused sim + top-k: one wave owns 16 rows; stream 16-col tiles of
// sim = ns @ ns^T via WMMA (two independent accumulator chains over K);
// per-row top-5 maintained in registers via LDS tile.
// ---------------------------------------------------------------------------
__global__ __launch_bounds__(32)
void topk_sim_kernel(const __bf16* __restrict__ ns, int* __restrict__ topk) {
  const int lane = threadIdx.x;
  const int m0 = blockIdx.x * 16;
  v16bf afr[8];
#pragma unroll
  for (int kk = 0; kk < 8; ++kk)
    afr[kk] = load_fragA(ns + (size_t)m0 * DIM + kk * 32, DIM, lane);
  __shared__ float Ssh[16][16];
  float tv[KTOP];
  int   ti[KTOP];
#pragma unroll
  for (int q = 0; q < KTOP; ++q) { tv[q] = -1e30f; ti[q] = 0; }
  const v8f z = {0.f, 0.f, 0.f, 0.f, 0.f, 0.f, 0.f, 0.f};
  for (int jt = 0; jt < NAGENT / 16; ++jt) {
    const int n0 = jt * 16;
    v8f acc0 = z, acc1 = z;  // two independent WMMA chains
#pragma unroll
    for (int kk = 0; kk < 4; ++kk) {
      v16bf b0 = load_fragB(ns + (size_t)n0 * DIM + kk * 32, DIM, lane);
      v16bf b1 = load_fragB(ns + (size_t)n0 * DIM + (kk + 4) * 32, DIM, lane);
      acc0 = wmma_bf16(afr[kk], b0, acc0);
      acc1 = wmma_bf16(afr[kk + 4], b1, acc1);
    }
    const int hi = lane >> 4, nn = lane & 15;
#pragma unroll
    for (int r = 0; r < 8; ++r) Ssh[r + hi * 8][nn] = acc0[r] + acc1[r];
    __syncthreads();
    if (lane < 16) {
      for (int c = 0; c < 16; ++c) {
        const float v = Ssh[lane][c];
        if (v > tv[KTOP - 1]) {
          int p = KTOP - 1;
          while (p > 0 && v > tv[p - 1]) {
            tv[p] = tv[p - 1]; ti[p] = ti[p - 1]; --p;
          }
          tv[p] = v; ti[p] = n0 + c;
        }
      }
    }
    __syncthreads();
  }
  if (lane < 16) {
#pragma unroll
    for (int q = 0; q < KTOP; ++q) topk[(m0 + lane) * KTOP + q] = ti[q];
  }
}

// ---------------------------------------------------------------------------
// Symmetrized adjacency as a bitset (N x N/32 words) + self loops.
// ---------------------------------------------------------------------------
__global__ void adj_build_kernel(const int* __restrict__ topk,
                                 unsigned* __restrict__ adj) {
  const int idx = blockIdx.x * blockDim.x + threadIdx.x;
  if (idx >= NAGENT * (KTOP + 1)) return;
  const int i = idx / (KTOP + 1), k = idx % (KTOP + 1);
  if (k == KTOP) {  // self loop (eye)
    atomicOr(&adj[(size_t)i * (NAGENT / 32) + (i >> 5)], 1u << (i & 31));
    return;
  }
  const int j = topk[i * KTOP + k];
  atomicOr(&adj[(size_t)i * (NAGENT / 32) + (j >> 5)], 1u << (j & 31));
  atomicOr(&adj[(size_t)j * (NAGENT / 32) + (i >> 5)], 1u << (i & 31));
}

// ---------------------------------------------------------------------------
// a_s[i,h] = <xs[i,h,:], att_src[h]>, a_d likewise.
// ---------------------------------------------------------------------------
__global__ void att_dots_kernel(const float* __restrict__ xs,
                                const float* __restrict__ att_src,
                                const float* __restrict__ att_dst,
                                float* __restrict__ a_s, float* __restrict__ a_d) {
  const int idx = blockIdx.x * blockDim.x + threadIdx.x;
  if (idx >= NAGENT * NHEAD) return;
  const int i = idx >> 2, h = idx & 3;
  const float* xr = xs + (size_t)i * DIM + h * HDIM;
  float s1 = 0.f, s2 = 0.f;
  for (int c = 0; c < HDIM; ++c) {
    const float v = xr[c];
    s1 += v * att_src[h * HDIM + c];
    s2 += v * att_dst[h * HDIM + c];
  }
  a_s[idx] = s1;
  a_d[idx] = s2;
}

// ---------------------------------------------------------------------------
// GAT aggregate per destination: sparse masked softmax over in-neighbors
// (non-edges contribute exp(-1e9-max)==0 exactly), then + bias, LayerNorm,
// + residual. Writes f32 and bf16 activations.
// ---------------------------------------------------------------------------
__global__ __launch_bounds__(128)
void gat_aggregate_kernel(const unsigned* __restrict__ adj,
                          const float* __restrict__ xs,
                          const float* __restrict__ a_s,
                          const float* __restrict__ a_d,
                          const float* __restrict__ bias,
                          const float* __restrict__ gamma,
                          const float* __restrict__ beta,
                          const float* __restrict__ x_in,
                          float* __restrict__ x_out,
                          __bf16* __restrict__ x_out_bf) {
  const int j = blockIdx.x, tid = threadIdx.x;
  __shared__ int nbr[NBR_CAP];
  __shared__ int cnt;
  __shared__ float alpha[NBR_CAP][NHEAD];
  __shared__ float red[128];
  __shared__ float hmax[NHEAD], hsum[NHEAD];
  __shared__ float mu_s, var_s;
  if (tid == 0) cnt = 0;
  __syncthreads();
  const unsigned* row = adj + (size_t)j * (NAGENT / 32);
  {
    unsigned bits = row[tid];  // 128 words, 128 threads
    while (bits) {
      const int b = __builtin_ctz(bits);
      bits &= bits - 1;
      const int p = atomicAdd(&cnt, 1);
      if (p < NBR_CAP) nbr[p] = tid * 32 + b;
    }
  }
  __syncthreads();
  const int n = cnt < NBR_CAP ? cnt : NBR_CAP;
  for (int h = 0; h < NHEAD; ++h) {
    const float adj_d = a_d[j * NHEAD + h];
    float lm = -1e30f;
    for (int p = tid; p < n; p += 128) {
      float l = a_s[nbr[p] * NHEAD + h] + adj_d;
      l = l > 0.f ? l : 0.2f * l;  // leaky relu
      lm = fmaxf(lm, l);
    }
    red[tid] = lm; __syncthreads();
    for (int s = 64; s > 0; s >>= 1) {
      if (tid < s) red[tid] = fmaxf(red[tid], red[tid + s]);
      __syncthreads();
    }
    if (tid == 0) hmax[h] = red[0];
    __syncthreads();
    float ls = 0.f;
    for (int p = tid; p < n; p += 128) {
      float l = a_s[nbr[p] * NHEAD + h] + adj_d;
      l = l > 0.f ? l : 0.2f * l;
      const float e = __expf(l - hmax[h]);
      alpha[p][h] = e;
      ls += e;
    }
    red[tid] = ls; __syncthreads();
    for (int s = 64; s > 0; s >>= 1) {
      if (tid < s) red[tid] += red[tid + s];
      __syncthreads();
    }
    if (tid == 0) hsum[h] = red[0];
    __syncthreads();
  }
  for (int p = tid; p < n; p += 128)
#pragma unroll
    for (int h = 0; h < NHEAD; ++h) alpha[p][h] /= hsum[h];
  __syncthreads();
  // each thread owns channels tid and tid+128
  float outv[2];
#pragma unroll
  for (int cc = 0; cc < 2; ++cc) {
    const int c = tid + cc * 128;
    const int h = c >> 6;
    float acc = 0.f;
    for (int p = 0; p < n; ++p) acc += alpha[p][h] * xs[(size_t)nbr[p] * DIM + c];
    outv[cc] = acc + bias[c];
  }
  // LayerNorm over 256 channels
  red[tid] = outv[0] + outv[1]; __syncthreads();
  for (int s = 64; s > 0; s >>= 1) { if (tid < s) red[tid] += red[tid + s]; __syncthreads(); }
  if (tid == 0) mu_s = red[0] / 256.f;
  __syncthreads();
  const float mu = mu_s;
  red[tid] = (outv[0] - mu) * (outv[0] - mu) + (outv[1] - mu) * (outv[1] - mu);
  __syncthreads();
  for (int s = 64; s > 0; s >>= 1) { if (tid < s) red[tid] += red[tid + s]; __syncthreads(); }
  if (tid == 0) var_s = red[0] / 256.f;
  __syncthreads();
  const float rstd = rsqrtf(var_s + 1e-5f);
#pragma unroll
  for (int cc = 0; cc < 2; ++cc) {
    const int c = tid + cc * 128;
    const float v =
        gamma[c] * (outv[cc] - mu) * rstd + beta[c] + x_in[(size_t)j * DIM + c];
    x_out[(size_t)j * DIM + c] = v;
    x_out_bf[(size_t)j * DIM + c] = f2bf(v);
  }
}

// ---------------------------------------------------------------------------
// Flash attention: one wave per (head, 16 queries). Stream 32-key tiles:
// S = QK^T (WMMA), online softmax (LDS row stats), ctx += P@V (WMMA).
// ---------------------------------------------------------------------------
__global__ __launch_bounds__(32)
void flash_attn_kernel(const __bf16* __restrict__ qkv, float* __restrict__ ctx,
                       __bf16* __restrict__ ctxbf) {
  const int lane = threadIdx.x;
  const int h = blockIdx.y;
  const int m0 = blockIdx.x * 16;
  const int ld = 3 * DIM;
  const int hi = lane >> 4, nn = lane & 15;
  v16bf qfr[2];
#pragma unroll
  for (int kk = 0; kk < 2; ++kk)
    qfr[kk] = load_fragA(qkv + (size_t)m0 * ld + h * HDIM + kk * 32, ld, lane);
  const v8f z = {0.f, 0.f, 0.f, 0.f, 0.f, 0.f, 0.f, 0.f};
  v8f cacc[4] = {z, z, z, z};
  __shared__ float Ssh[16][32];
  __shared__ __align__(16) __bf16 Vsh[32][HDIM];
  __shared__ float rcorr[16], rlsh[16];
  float rm = -1e30f, rl = 0.f;
  for (int jt = 0; jt < NAGENT / 32; ++jt) {
    const int n0 = jt * 32;
    // S = Q K^T over 32 keys (two 16-wide N tiles = 2 independent WMMA chains)
    v16bf kb[4];
#pragma unroll
    for (int half = 0; half < 2; ++half)
#pragma unroll
      for (int kk = 0; kk < 2; ++kk)
        kb[half * 2 + kk] = load_fragB(
            qkv + (size_t)(n0 + half * 16) * ld + DIM + h * HDIM + kk * 32, ld, lane);
    v8f s0 = z, s1 = z;
    s0 = wmma_bf16(qfr[0], kb[0], s0);
    s1 = wmma_bf16(qfr[0], kb[2], s1);
    s0 = wmma_bf16(qfr[1], kb[1], s0);
    s1 = wmma_bf16(qfr[1], kb[3], s1);
#pragma unroll
    for (int r = 0; r < 8; ++r) {
      Ssh[r + hi * 8][nn]      = s0[r] * 0.125f;  // 1/sqrt(64)
      Ssh[r + hi * 8][16 + nn] = s1[r] * 0.125f;
    }
    // stage V tile (32 keys x 64) into LDS; lane copies its key row
#pragma unroll
    for (int t = 0; t < HDIM / 8; ++t)
      *(v8bf*)&Vsh[lane][t * 8] =
          *(const v8bf*)(qkv + (size_t)(n0 + lane) * ld + 2 * DIM + h * HDIM + t * 8);
    __syncthreads();
    if (lane < 16) {  // online softmax row update
      float tmax = -1e30f;
      for (int c = 0; c < 32; ++c) tmax = fmaxf(tmax, Ssh[lane][c]);
      const float nm = fmaxf(rm, tmax);
      const float corr = __expf(rm - nm);
      float nl = rl * corr;
      for (int c = 0; c < 32; ++c) {
        const float e = __expf(Ssh[lane][c] - nm);
        Ssh[lane][c] = e;
        nl += e;
      }
      rm = nm; rl = nl;
      rcorr[lane] = corr;
    }
    __syncthreads();
#pragma unroll
    for (int r = 0; r < 8; ++r) {
      const float cr = rcorr[r + hi * 8];
#pragma unroll
      for (int t = 0; t < 4; ++t) cacc[t][r] *= cr;
    }
    // P fragment (16x32, A layout) from exponentiated scores
    v16bf p;
#pragma unroll
    for (int e = 0; e < 8; ++e) {
      p[e]     = f2bf(Ssh[nn][hi * 16 + e]);
      p[e + 8] = f2bf(Ssh[nn][hi * 16 + 8 + e]);
    }
    // ctx += P @ V  (4 independent column tiles of hd=64)
#pragma unroll
    for (int t = 0; t < 4; ++t) {
      v16bf b;
      const int c = t * 16 + nn;
#pragma unroll
      for (int e = 0; e < 16; ++e) b[e] = Vsh[hi * 16 + e][c];
      cacc[t] = wmma_bf16(p, b, cacc[t]);
    }
    __syncthreads();
  }
  if (lane < 16) rlsh[lane] = rl;
  __syncthreads();
#pragma unroll
  for (int r = 0; r < 8; ++r) {
    const int row = m0 + r + hi * 8;
    const float inv = 1.f / rlsh[r + hi * 8];
#pragma unroll
    for (int t = 0; t < 4; ++t) {
      const int col = h * HDIM + t * 16 + nn;
      const float v = cacc[t][r] * inv;
      ctx[(size_t)row * DIM + col] = v;
      ctxbf[(size_t)row * DIM + col] = f2bf(v);
    }
  }
}

// ---------------------------------------------------------------------------
// gate_in = [states_bf | agg_bf]
// ---------------------------------------------------------------------------
__global__ void concat_kernel(const __bf16* __restrict__ a,
                              const __bf16* __restrict__ b,
                              __bf16* __restrict__ dst) {
  const int idx = blockIdx.x * blockDim.x + threadIdx.x;
  if (idx >= NAGENT * 2 * DIM) return;
  const int i = idx >> 9, c = idx & 511;
  dst[idx] = (c < DIM) ? a[(size_t)i * DIM + c] : b[(size_t)i * DIM + (c - DIM)];
}

// ---------------------------------------------------------------------------
// strength = sigmoid(hidden @ gate_W2 + b2); gated = bf16(agg * strength)
// ---------------------------------------------------------------------------
__global__ __launch_bounds__(256)
void gate_kernel(const float* __restrict__ hidden, const float* __restrict__ W2,
                 const float* __restrict__ b2, const float* __restrict__ agg,
                 __bf16* __restrict__ gated) {
  const int i = blockIdx.x, c = threadIdx.x;
  __shared__ float red[256];
  red[c] = hidden[(size_t)i * DIM + c] * W2[c];
  __syncthreads();
  for (int s = 128; s > 0; s >>= 1) { if (c < s) red[c] += red[c + s]; __syncthreads(); }
  const float strength = 1.f / (1.f + __expf(-(red[0] + b2[0])));
  gated[(size_t)i * DIM + c] = f2bf(agg[(size_t)i * DIM + c] * strength);
}

// ---------------------------------------------------------------------------
// host side
// ---------------------------------------------------------------------------
static void gemm_launch(hipStream_t s, const __bf16* A, int lda, const __bf16* Bt,
                        int ldb, float* C, __bf16* Cbf, int ldc, int M, int Nc,
                        int K, const float* bias, const float* residual, int flags) {
  if (Nc % 64 == 0) {
    dim3 grid(Nc / 64, M / 16);
    gemm_bf16_wmma_kernel<4><<<grid, 32, 0, s>>>(A, lda, Bt, ldb, C, Cbf, ldc, K,
                                                 bias, residual, flags);
  } else {  // Nc multiple of 32 (message GEMM)
    dim3 grid(Nc / 32, M / 16);
    gemm_bf16_wmma_kernel<2><<<grid, 32, 0, s>>>(A, lda, Bt, ldb, C, Cbf, ldc, K,
                                                 bias, residual, flags);
  }
}

extern "C" void kernel_launch(void* const* d_in, const int* in_sizes, int n_in,
                              void* d_out, int out_size, void* d_ws, size_t ws_size,
                              hipStream_t stream) {
  (void)in_sizes; (void)n_in; (void)out_size; (void)ws_size;
  const int N = NAGENT, D = DIM;

  const float* states    = (const float*)d_in[0];
  const float* agent_emb = (const float*)d_in[1];
  const float* role_emb  = (const float*)d_in[2];
  const float* gat_W     = (const float*)d_in[3];
  const float* att_src   = (const float*)d_in[4];
  const float* att_dst   = (const float*)d_in[5];
  const float* gat_bias  = (const float*)d_in[6];
  const float* ln_gamma  = (const float*)d_in[7];
  const float* ln_beta   = (const float*)d_in[8];
  const float* enc_W1 = (const float*)d_in[9];   const float* enc_b1 = (const float*)d_in[10];
  const float* enc_W2 = (const float*)d_in[11];  const float* enc_b2 = (const float*)d_in[12];
  const float* dec_W1 = (const float*)d_in[13];  const float* dec_b1 = (const float*)d_in[14];
  const float* dec_W2 = (const float*)d_in[15];  const float* dec_b2 = (const float*)d_in[16];
  const float* mha_in_w  = (const float*)d_in[17]; const float* mha_in_b  = (const float*)d_in[18];
  const float* mha_out_w = (const float*)d_in[19]; const float* mha_out_b = (const float*)d_in[20];
  const float* proj_W = (const float*)d_in[21];  const float* proj_b = (const float*)d_in[22];
  const float* gate_W1 = (const float*)d_in[23]; const float* gate_b1 = (const float*)d_in[24];
  const float* gate_W2 = (const float*)d_in[25]; const float* gate_b2 = (const float*)d_in[26];

  char* ws = (char*)d_ws;
  size_t off = 0;
  auto alloc = [&](size_t bytes) -> void* {
    void* p = ws + off;
    off = (off + bytes + 255) & ~(size_t)255;
    return p;
  };
  const size_t ND = (size_t)N * D;
  float*  xA       = (float*)alloc(ND * 4);
  __bf16* xA_bf    = (__bf16*)alloc(ND * 2);
  float*  xB       = (float*)alloc(ND * 4);
  __bf16* xB_bf    = (__bf16*)alloc(ND * 2);
  __bf16* ns_bf    = (__bf16*)alloc(ND * 2);
  __bf16* states_bf= (__bf16*)alloc(ND * 2);
  float*  xs       = (float*)alloc(ND * 4);
  float*  a_s      = (float*)alloc((size_t)N * NHEAD * 4);
  float*  a_d      = (float*)alloc((size_t)N * NHEAD * 4);
  int*    topk     = (int*)alloc((size_t)N * KTOP * 4);
  unsigned* adj    = (unsigned*)alloc((size_t)N * (N / 32) * 4);
  __bf16* gat_Wt   = (__bf16*)alloc((size_t)2 * D * D * 2);
  __bf16* encW1t   = (__bf16*)alloc((size_t)64 * 256 * 2);
  __bf16* encW2t   = (__bf16*)alloc((size_t)32 * 64 * 2);
  __bf16* decW1t   = (__bf16*)alloc((size_t)128 * 32 * 2);
  __bf16* decW2t   = (__bf16*)alloc((size_t)256 * 128 * 2);
  __bf16* mhaInWt  = (__bf16*)alloc((size_t)768 * 256 * 2);
  __bf16* mhaOutWt = (__bf16*)alloc((size_t)256 * 256 * 2);
  __bf16* gateW1t  = (__bf16*)alloc((size_t)256 * 512 * 2);
  __bf16* projWt   = (__bf16*)alloc((size_t)256 * 256 * 2);
  float*  h1       = (float*)alloc((size_t)N * 64 * 4);
  __bf16* h1_bf    = (__bf16*)alloc((size_t)N * 64 * 2);
  float*  msg      = (float*)alloc((size_t)N * 32 * 4);
  __bf16* msg_bf   = (__bf16*)alloc((size_t)N * 32 * 2);
  float*  dh       = (float*)alloc((size_t)N * 128 * 4);
  __bf16* dh_bf    = (__bf16*)alloc((size_t)N * 128 * 2);
  float*  decf     = (float*)alloc(ND * 4);
  __bf16* dec_bf   = (__bf16*)alloc(ND * 2);
  float*  qkv      = (float*)alloc((size_t)N * 768 * 4);
  __bf16* qkv_bf   = (__bf16*)alloc((size_t)N * 768 * 2);
  float*  ctx      = (float*)alloc(ND * 4);
  __bf16* ctx_bf   = (__bf16*)alloc(ND * 2);
  float*  agg      = (float*)alloc(ND * 4);
  __bf16* agg_bf   = (__bf16*)alloc(ND * 2);
  __bf16* gatein_bf= (__bf16*)alloc((size_t)N * 512 * 2);
  float*  ghid     = (float*)alloc(ND * 4);
  __bf16* gated_bf = (__bf16*)alloc(ND * 2);

  // 1) prep
  prep_kernel<<<N, 256, 0, stream>>>(states, agent_emb, role_emb, xA, xA_bf, ns_bf,
                                     states_bf);

  // 2) weight transposes -> bf16 [N][K]
  auto tcast = [&](const float* src, __bf16* dst, int R, int C) {
    const int tot = R * C;
    transpose_cast_kernel<<<(tot + 255) / 256, 256, 0, stream>>>(src, dst, R, C);
  };
  tcast(gat_W, gat_Wt, D, D);
  tcast(gat_W + (size_t)D * D, gat_Wt + (size_t)D * D, D, D);
  tcast(enc_W1, encW1t, 256, 64);
  tcast(enc_W2, encW2t, 64, 32);
  tcast(dec_W1, decW1t, 32, 128);
  tcast(dec_W2, decW2t, 128, 256);
  tcast(mha_in_w, mhaInWt, 256, 768);
  tcast(mha_out_w, mhaOutWt, 256, 256);
  tcast(gate_W1, gateW1t, 512, 256);
  tcast(proj_W, projWt, 256, 256);

  // 3) fused sim + top-k, adjacency bitset
  topk_sim_kernel<<<N / 16, 32, 0, stream>>>(ns_bf, topk);
  hipMemsetAsync(adj, 0, (size_t)N * (N / 32) * 4, stream);
  adj_build_kernel<<<(N * (KTOP + 1) + 255) / 256, 256, 0, stream>>>(topk, adj);

  // 4) GAT layers
  float*  xcur = xA;      __bf16* xcur_bf = xA_bf;
  float*  xnxt = xB;      __bf16* xnxt_bf = xB_bf;
  for (int l = 0; l < 2; ++l) {
    gemm_launch(stream, xcur_bf, D, gat_Wt + (size_t)l * D * D, D, xs, nullptr, D,
                N, D, D, nullptr, nullptr, 0);
    att_dots_kernel<<<(N * NHEAD + 255) / 256, 256, 0, stream>>>(
        xs, att_src + (size_t)l * NHEAD * HDIM, att_dst + (size_t)l * NHEAD * HDIM,
        a_s, a_d);
    gat_aggregate_kernel<<<N, 128, 0, stream>>>(
        adj, xs, a_s, a_d, gat_bias + (size_t)l * D, ln_gamma + (size_t)l * D,
        ln_beta + (size_t)l * D, xcur, xnxt, xnxt_bf);
    float* tf = xcur; xcur = xnxt; xnxt = tf;
    __bf16* tb = xcur_bf; xcur_bf = xnxt_bf; xnxt_bf = tb;
  }

  // 5) message encode/decode MLPs
  gemm_launch(stream, xcur_bf, D, encW1t, 256, h1, h1_bf, 64, N, 64, 256, enc_b1,
              nullptr, 1);
  gemm_launch(stream, h1_bf, 64, encW2t, 64, msg, msg_bf, 32, N, 32, 64, enc_b2,
              nullptr, 0);
  gemm_launch(stream, msg_bf, 32, decW1t, 32, dh, dh_bf, 128, N, 128, 32, dec_b1,
              nullptr, 1);
  gemm_launch(stream, dh_bf, 128, decW2t, 128, decf, dec_bf, 256, N, 256, 128,
              dec_b2, nullptr, 0);

  // 6) qkv projection + flash attention + output projection
  gemm_launch(stream, dec_bf, 256, mhaInWt, 256, qkv, qkv_bf, 768, N, 768, 256,
              mha_in_b, nullptr, 0);
  {
    dim3 grid(N / 16, NHEAD);
    flash_attn_kernel<<<grid, 32, 0, stream>>>(qkv_bf, ctx, ctx_bf);
  }
  gemm_launch(stream, ctx_bf, 256, mhaOutWt, 256, agg, agg_bf, 256, N, 256, 256,
              mha_out_b, nullptr, 0);

  // 7) gate + final projection with residual
  concat_kernel<<<(N * 512 + 255) / 256, 256, 0, stream>>>(states_bf, agg_bf,
                                                           gatein_bf);
  gemm_launch(stream, gatein_bf, 512, gateW1t, 512, ghid, nullptr, 256, N, 256, 512,
              gate_b1, nullptr, 1);
  gate_kernel<<<N, 256, 0, stream>>>(ghid, gate_W2, gate_b2, agg, gated_bf);
  gemm_launch(stream, gated_bf, 256, projWt, 256, (float*)d_out, nullptr, 256, N,
              256, 256, proj_b, states, 0);
}
